// AttentionLayer_28587302322788
// MI455X (gfx1250) — matrix operand order repaired and checked by reading, not compiled, not run on hardware
//
#include <hip/hip_runtime.h>

// ---------------------------------------------------------------------------
// CDNA5 (gfx1250) transformer block: QKV proj -> causal MHA -> Wo -> LN ->
// FFN(relu) -> LN -> final linear.  All GEMM-shaped math via
// v_wmma_f32_16x16x32_bf16 (wave32).
// ---------------------------------------------------------------------------

typedef __bf16 bf16;
typedef __attribute__((ext_vector_type(16))) __bf16 v16bf;
typedef __attribute__((ext_vector_type(8)))  __bf16 v8bf;
typedef __attribute__((ext_vector_type(8)))  float  v8f;

#define BATCH   4
#define SEQ     2048
#define DMODEL  512
#define NHEADS  8
#define HEAD    64
#define MTOK    (BATCH * SEQ)   // 8192

__device__ __forceinline__ v8f wmma_bf16(v16bf a, v16bf b, v8f c) {
  // 8 args: (neg_a, A, neg_b, B, c_mod, C, reuse_a, reuse_b)
  return __builtin_amdgcn_wmma_f32_16x16x32_bf16(false, a, false, b,
                                                 (short)0, c, false, false);
}

__device__ __forceinline__ v16bf cat8(v8bf lo, v8bf hi) {
  v16bf r;
#pragma unroll
  for (int i = 0; i < 8; ++i) { r[i] = lo[i]; r[i + 8] = hi[i]; }
  return r;
}

// ---------------------------------------------------------------------------
// GEMM: Y[M,N] = act(X[M,K] @ W[N,K]^T + bias[N])
// Block = 256 threads (8 waves). Tile 128(M) x 128(N), K step = 32.
// Wave tile 32(M) x 64(N) = 2x4 WMMA 16x16 C tiles (8 WMMA / K-step).
// Register + LDS double buffering: next K-slab is pulled into VGPRs before
// the WMMAs, written to the alternate LDS buffer after, 1 barrier / K-step.
// M % 128 == 0, N % 128 == 0, K % 32 == 0 for all problem sizes here.
// ---------------------------------------------------------------------------
template <bool RELU, bool OUT_BF16>
__global__ __launch_bounds__(256) void gemm_bias_kernel(
    const float* __restrict__ X, const float* __restrict__ W,
    const float* __restrict__ bias, void* __restrict__ Yv,
    int M, int N, int K) {
  // padded LDS rows: 40 bf16 = 80 bytes (16B-aligned rows), 2 buffers
  __shared__ bf16 Xs[2][128 * 40];
  __shared__ bf16 Ws[2][128 * 40];

  const int tid  = threadIdx.x;
  const int lane = tid & 31;
  const int wid  = tid >> 5;
  const int wm   = wid >> 1;    // 0..3 (32 rows each)
  const int wn   = wid & 1;     // 0..1 (64 cols each)
  const int lr   = lane & 15;
  const int hi   = lane >> 4;   // 0 or 1

  const int n0 = blockIdx.x * 128;
  const int m0 = blockIdx.y * 128;

  // staging coordinates: thread covers rows r0, r0+32, r0+64, r0+96 at quad cq
  const int r0 = tid >> 3;      // 0..31
  const int cq = tid & 7;       // quad column (4 floats)

  float4 xr[4], wr[4];
  auto load_stage = [&](int k0) {
#pragma unroll
    for (int it = 0; it < 4; ++it) {
      const int row = r0 + it * 32;
      xr[it] = *(const float4*)(X + (size_t)(m0 + row) * K + k0 + cq * 4);
      wr[it] = *(const float4*)(W + (size_t)(n0 + row) * K + k0 + cq * 4);
    }
  };
  auto store_stage = [&](int buf) {
#pragma unroll
    for (int it = 0; it < 4; ++it) {
      const int row = r0 + it * 32;
      bf16* px = &Xs[buf][row * 40 + cq * 4];
      px[0] = (bf16)xr[it].x; px[1] = (bf16)xr[it].y;
      px[2] = (bf16)xr[it].z; px[3] = (bf16)xr[it].w;
      bf16* pw = &Ws[buf][row * 40 + cq * 4];
      pw[0] = (bf16)wr[it].x; pw[1] = (bf16)wr[it].y;
      pw[2] = (bf16)wr[it].z; pw[3] = (bf16)wr[it].w;
    }
  };

  v8f c[2][4] = {};

  load_stage(0);
  store_stage(0);
  __syncthreads();

  for (int k0 = 0; k0 < K; k0 += 32) {
    const int  cur  = (k0 >> 5) & 1;
    const bool more = (k0 + 32) < K;
    if (more) load_stage(k0 + 32);   // global loads in flight under the WMMAs

    // A fragments (16x32): lanes 0-15 row m, slots 0..7 = k0..7, 8..15 = k16..23
    //                      lanes 16-31 row m, slots 0..7 = k8..15, 8..15 = k24..31
    v16bf Af[2], Bf[4];
#pragma unroll
    for (int tm = 0; tm < 2; ++tm) {
      const bf16* r = &Xs[cur][(wm * 32 + tm * 16 + lr) * 40];
      Af[tm] = cat8(*(const v8bf*)(r + hi * 8), *(const v8bf*)(r + 16 + hi * 8));
    }
    // B fragments (32x16): lane n col n, same k order as A (k0..7,16..23 / k8..15,24..31)
#pragma unroll
    for (int tn = 0; tn < 4; ++tn) {
      const bf16* r = &Ws[cur][(wn * 64 + tn * 16 + lr) * 40];
      Bf[tn] = cat8(*(const v8bf*)(r + hi * 8), *(const v8bf*)(r + 16 + hi * 8));
    }
#pragma unroll
    for (int tm = 0; tm < 2; ++tm)
#pragma unroll
      for (int tn = 0; tn < 4; ++tn)
        c[tm][tn] = wmma_bf16(Af[tm], Bf[tn], c[tm][tn]);

    if (more) store_stage(cur ^ 1);  // write-only to the other buffer: safe
    __syncthreads();
  }

  // ---- epilogue: C tile: VGPR r -> row r (lanes 0-15) / r+8 (16-31),
  //      col = lane&15 ----
#pragma unroll
  for (int tm = 0; tm < 2; ++tm) {
#pragma unroll
    for (int tn = 0; tn < 4; ++tn) {
      const int col = n0 + wn * 64 + tn * 16 + lr;
      const float bb = bias[col];
#pragma unroll
      for (int r = 0; r < 8; ++r) {
        const int row = m0 + wm * 32 + tm * 16 + r + 8 * hi;
        float vv = c[tm][tn][r] + bb;
        if (RELU) vv = fmaxf(vv, 0.0f);
        if (OUT_BF16)
          ((bf16*)Yv)[(size_t)row * N + col] = (bf16)vv;
        else
          ((float*)Yv)[(size_t)row * N + col] = vv;
      }
    }
  }
}

// ---------------------------------------------------------------------------
// Causal flash attention (bf16 WMMA, f32 accumulate).
// One wave = one 16-query block; 4 waves / block share the same (b, h) and
// cooperatively double-buffer V^T tiles in LDS (4x less global V traffic).
// Scores computed transposed: S^T = K_blk(16xE) @ Q^T(Ex16) so per-query
// softmax reductions are one shfl_xor(16), and the P^T B-fragment needs only
// a cross-half shuffle.
// ---------------------------------------------------------------------------
__global__ __launch_bounds__(128) void attn_kernel(
    const bf16* __restrict__ Qp, const bf16* __restrict__ Kp,
    const bf16* __restrict__ Vp, float* __restrict__ att) {
  __shared__ bf16 vlds[2][64 * 40];   // V^T [e][s], padded rows of 40, 2 bufs

  const int tidb = threadIdx.x;       // 0..127
  const int lane = tidb & 31;
  const int wid  = tidb >> 5;

  const int gid = blockIdx.x * 4 + wid;     // 0..4095
  const int qb  = gid & 127;                // query block (16 rows)
  const int h   = (gid >> 7) & 7;           // uniform across block
  const int b   = gid >> 10;                // uniform across block

  const int lr = lane & 15;
  const int hi = lane >> 4;
  const size_t base = (size_t)b * SEQ * DMODEL;

  // Q^T B-fragments (E is the contraction dim): lane n = query col lr.
  const bf16* qrow = Qp + base + (size_t)(qb * 16 + lr) * DMODEL + h * HEAD;
  const v16bf Bq0 = cat8(*(const v8bf*)(qrow + hi * 8),
                         *(const v8bf*)(qrow + 16 + hi * 8));
  const v16bf Bq1 = cat8(*(const v8bf*)(qrow + 32 + hi * 8),
                         *(const v8bf*)(qrow + 48 + hi * 8));

  float mrow = -3.0e38f;   // running max for query q = lane&15
  float srow = 0.0f;       // running sum
  v8f o[4] = {};           // O^T accumulators: 4 E-tiles x 16 queries

  const int q_global = qb * 16 + lr;
  const int nkb = (qb * 16 + 15) / 32 + 1;          // this wave's causal bound
  const int qb0 = (blockIdx.x * 4) & 127;           // block's first qb
  const int nkb_max = ((qb0 + 3) * 16 + 15) / 32 + 1;

  for (int kb = 0; kb < nkb_max; ++kb) {
    const int s0 = kb * 32;
    bf16* vbuf = vlds[kb & 1];

    // ---- cooperative stage of V^T (32 s x 64 e), all 128 threads ----
#pragma unroll
    for (int it = 0; it < 8; ++it) {
      const int idx = tidb + it * 128;   // 0..1023
      const int s   = idx >> 5;          // 0..31
      const int ep  = idx & 31;          // e pair
      const bf16* vr =
          Vp + base + (size_t)(s0 + s) * DMODEL + h * HEAD + 2 * ep;
      vbuf[(2 * ep) * 40 + s]     = vr[0];
      vbuf[(2 * ep + 1) * 40 + s] = vr[1];
    }
    __syncthreads();   // one barrier per key block (double-buffered)

    if (kb < nkb) {
      // ---- scores^T: two 16(key)x16(q) tiles ----
      v8f sc[2];
#pragma unroll
      for (int kt = 0; kt < 2; ++kt) {
        const bf16* krow =
            Kp + base + (size_t)(s0 + kt * 16 + lr) * DMODEL + h * HEAD;
        const v16bf Ak0 = cat8(*(const v8bf*)(krow + hi * 8),
                               *(const v8bf*)(krow + 16 + hi * 8));
        const v16bf Ak1 = cat8(*(const v8bf*)(krow + 32 + hi * 8),
                               *(const v8bf*)(krow + 48 + hi * 8));
        v8f t = {};
        t = wmma_bf16(Ak0, Bq0, t);
        t = wmma_bf16(Ak1, Bq1, t);
        sc[kt] = t;
      }

      // ---- scale + causal mask + online softmax ----
      const float scale = 0.125f;   // 1/sqrt(64)
      float loc[16];
#pragma unroll
      for (int r = 0; r < 8; ++r) {
        const int key0 = s0 + r + 8 * hi;          // tile0 key
        const int key1 = s0 + 16 + r + 8 * hi;     // tile1 key
        float v0 = sc[0][r] * scale; if (key0 > q_global) v0 = -1.0e30f;
        float v1 = sc[1][r] * scale; if (key1 > q_global) v1 = -1.0e30f;
        loc[r] = v0; loc[8 + r] = v1;
      }
      float bm = loc[0];
#pragma unroll
      for (int i = 1; i < 16; ++i) bm = fmaxf(bm, loc[i]);
      bm = fmaxf(bm, __shfl_xor(bm, 16));
      const float mnew  = fmaxf(mrow, bm);
      const float alpha = __expf(mrow - mnew);
      float ps = 0.0f;
#pragma unroll
      for (int i = 0; i < 16; ++i) {
        loc[i] = __expf(loc[i] - mnew);
        ps += loc[i];
      }
      ps += __shfl_xor(ps, 16);
      srow = srow * alpha + ps;
      mrow = mnew;
#pragma unroll
      for (int t = 0; t < 4; ++t)
#pragma unroll
        for (int r = 0; r < 8; ++r) o[t][r] *= alpha;

      // ---- build P^T B-fragment (32 s x 16 q) via cross-half shuffles ----
      v16bf Bp;
#pragma unroll
      for (int r = 0; r < 8; ++r) {
        const float o0 = __shfl_xor(loc[r], 16);       // tile0, other half
        const float o1 = __shfl_xor(loc[8 + r], 16);   // tile1, other half
        const float s_lo = hi ? o1 : loc[r];           // slots 0..7
        const float s_hi = hi ? loc[8 + r] : o0;       // slots 8..15
        Bp[r]     = (bf16)s_lo;
        Bp[8 + r] = (bf16)s_hi;
      }

      // ---- O^T += V^T(Ex32) @ P^T(32x16q): 4 A-fragments from LDS ----
#pragma unroll
      for (int t = 0; t < 4; ++t) {
        const bf16* er = &vbuf[(t * 16 + lr) * 40];
        const v16bf Av = cat8(*(const v8bf*)(er + hi * 16),
                              *(const v8bf*)(er + hi * 16 + 8));
        o[t] = wmma_bf16(Av, Bp, o[t]);
      }
    }
  }

  // ---- write O / sum: element (e = t*16 + r + 8*hi, q = lane&15) ----
  const float inv = 1.0f / srow;
  float* orow = att + base + (size_t)(qb * 16 + lr) * DMODEL + h * HEAD;
#pragma unroll
  for (int t = 0; t < 4; ++t)
#pragma unroll
    for (int r = 0; r < 8; ++r)
      orow[t * 16 + r + 8 * hi] = o[t][r] * inv;
}

// ---------------------------------------------------------------------------
// out = LayerNorm(a + b) * g + beta     (row width = 512, one block per row)
// ---------------------------------------------------------------------------
__global__ __launch_bounds__(256) void resid_ln_kernel(
    const float* __restrict__ A, const float* __restrict__ Bm,
    const float* __restrict__ g, const float* __restrict__ be,
    float* __restrict__ out) {
  const int row = blockIdx.x;
  const int tid = threadIdx.x;
  const float* pa = A  + (size_t)row * DMODEL;
  const float* pb = Bm + (size_t)row * DMODEL;

  const float x0 = pa[tid] + pb[tid];
  const float x1 = pa[tid + 256] + pb[tid + 256];
  float s = x0 + x1, q2 = x0 * x0 + x1 * x1;
#pragma unroll
  for (int off = 16; off >= 1; off >>= 1) {
    s  += __shfl_xor(s, off);
    q2 += __shfl_xor(q2, off);
  }
  __shared__ float rs[8], rq[8];
  const int lane = tid & 31, wid = tid >> 5;
  if (lane == 0) { rs[wid] = s; rq[wid] = q2; }
  __syncthreads();
  if (tid == 0) {
    float ts = 0.0f, tq = 0.0f;
#pragma unroll
    for (int i = 0; i < 8; ++i) { ts += rs[i]; tq += rq[i]; }
    rs[0] = ts; rq[0] = tq;
  }
  __syncthreads();
  const float mean = rs[0] * (1.0f / 512.0f);
  const float var  = rq[0] * (1.0f / 512.0f) - mean * mean;
  const float rstd = rsqrtf(var + 1e-5f);
  out[(size_t)row * DMODEL + tid]       = (x0 - mean) * rstd * g[tid] + be[tid];
  out[(size_t)row * DMODEL + tid + 256] =
      (x1 - mean) * rstd * g[tid + 256] + be[tid + 256];
}

// ---------------------------------------------------------------------------
extern "C" void kernel_launch(void* const* d_in, const int* in_sizes, int n_in,
                              void* d_out, int out_size, void* d_ws,
                              size_t ws_size, hipStream_t stream) {
  const float* q   = (const float*)d_in[0];
  const float* k   = (const float*)d_in[1];
  const float* v   = (const float*)d_in[2];
  const float* Wq  = (const float*)d_in[3];
  const float* bq  = (const float*)d_in[4];
  const float* Wk  = (const float*)d_in[5];
  const float* bk  = (const float*)d_in[6];
  const float* Wv  = (const float*)d_in[7];
  const float* bv  = (const float*)d_in[8];
  const float* Wo  = (const float*)d_in[9];
  const float* bo  = (const float*)d_in[10];
  const float* Wc1 = (const float*)d_in[11];
  const float* bc1 = (const float*)d_in[12];
  const float* Wc2 = (const float*)d_in[13];
  const float* bc2 = (const float*)d_in[14];
  const float* g1  = (const float*)d_in[15];
  const float* b1  = (const float*)d_in[16];
  const float* g2  = (const float*)d_in[17];
  const float* b2  = (const float*)d_in[18];
  const float* Wl  = (const float*)d_in[19];
  const float* bl  = (const float*)d_in[20];

  char* ws = (char*)d_ws;
  const size_t MB = 1024u * 1024u;
  // Liveness-based layout (peak 104 MB):
  bf16*  Qp   = (bf16*)(ws + 0);          //  8 MB
  bf16*  Kp   = (bf16*)(ws + 8 * MB);     //  8 MB
  bf16*  Vp   = (bf16*)(ws + 16 * MB);    //  8 MB
  float* att  = (float*)(ws + 24 * MB);   // 16 MB
  float* attO = (float*)(ws + 0);         // 16 MB (over Qp/Kp, dead)
  float* x    = (float*)(ws + 24 * MB);   // 16 MB (over att, dead)
  float* y1   = (float*)(ws + 40 * MB);   // 64 MB
  float* y2   = (float*)(ws + 0);         // 16 MB (over attO, dead)
  float* z    = (float*)(ws + 40 * MB);   // 16 MB (over y1, dead)
  float* out  = (float*)d_out;

  const dim3 blk(256);
  const dim3 g512(DMODEL / 128, MTOK / 128);   // N=512 GEMMs  (4, 64)
  const dim3 g2048(2048 / 128, MTOK / 128);    // N=2048 GEMM  (16, 64)

  // QKV projections (bf16 outputs feed WMMA attention directly)
  gemm_bias_kernel<false, true><<<g512, blk, 0, stream>>>(q, Wq, bq, (void*)Qp,
                                                          MTOK, DMODEL, DMODEL);
  gemm_bias_kernel<false, true><<<g512, blk, 0, stream>>>(k, Wk, bk, (void*)Kp,
                                                          MTOK, DMODEL, DMODEL);
  gemm_bias_kernel<false, true><<<g512, blk, 0, stream>>>(v, Wv, bv, (void*)Vp,
                                                          MTOK, DMODEL, DMODEL);
  // Causal MHA: 4096 waves, 4 per block sharing V staging
  attn_kernel<<<dim3(1024), dim3(128), 0, stream>>>(Qp, Kp, Vp, att);
  // Output projection
  gemm_bias_kernel<false, false><<<g512, blk, 0, stream>>>(
      att, Wo, bo, (void*)attO, MTOK, DMODEL, DMODEL);
  // x = LN(v + att@Wo^T)
  resid_ln_kernel<<<dim3(MTOK), blk, 0, stream>>>(v, attO, g1, b1, x);
  // FFN
  gemm_bias_kernel<true, false><<<g2048, blk, 0, stream>>>(
      x, Wc1, bc1, (void*)y1, MTOK, 2048, DMODEL);
  gemm_bias_kernel<false, false><<<g512, blk, 0, stream>>>(
      y1, Wc2, bc2, (void*)y2, MTOK, DMODEL, 2048);
  // z = LN(x + y)
  resid_ln_kernel<<<dim3(MTOK), blk, 0, stream>>>(x, y2, g2, b2, z);
  // final linear
  gemm_bias_kernel<false, false><<<g512, blk, 0, stream>>>(
      z, Wl, bl, (void*)out, MTOK, DMODEL, DMODEL);
}
